// BasicMixtureOfExperts_89223650607341
// MI455X (gfx1250) — matrix operand order repaired and checked by reading, not compiled, not run on hardware
//
#include <hip/hip_runtime.h>
#include <hip/hip_bf16.h>

// MoE top-2 routed FFN for MI455X (gfx1250, wave32).
// init -> gate(top2 + compaction + lb partials) -> finalize(prefix/lb)
//      -> FFN1 (bf16 WMMA 128x128 tiles, 2x-unrolled double-buffered LDS, relu, h->ws)
//      -> FFN2 (bf16 WMMA, same pipeline, atomic scatter-add).

#define NT   2048      // B*T tokens
#define DDIM 1024
#define HDIM 4096
#define NE   8
#define LDS_STRIDE 40          // 32 + 8 pad (bf16 elems); row pitch 80B keeps 16B alignment
#define TILE_E (128 * LDS_STRIDE)

typedef __attribute__((ext_vector_type(16))) __bf16 v16bf;
typedef __attribute__((ext_vector_type(2)))  __bf16 bf16x2;
typedef __attribute__((ext_vector_type(8)))  float  v8f;
typedef __attribute__((ext_vector_type(2)))  float  f32x2;
typedef __attribute__((ext_vector_type(4)))  float  f32x4;
typedef __attribute__((ext_vector_type(4)))  unsigned int u32x4;

// pack two f32 -> packed bf16 pair; convertvector lowers to two-source v_cvt_pk_bf16_f32
__device__ __forceinline__ unsigned pkbf(float a, float b) {
  f32x2 f; f[0] = a; f[1] = b;
  bf16x2 v = __builtin_convertvector(f, bf16x2);
  return __builtin_bit_cast(unsigned, v);
}

// ---- WMMA fragment loads from LDS (row-major tiles, stride LDS_STRIDE bf16) ----
// A (16x32, 16-bit): lane m (0-15) holds row m, K=0..15; lane m+16: K=16..31
__device__ __forceinline__ v16bf load_a_frag(const __bf16* tile, int mbase, int lane) {
  const int m  = mbase + (lane & 15);
  const int kh = lane >> 4;
  const __bf16* p = tile + m * LDS_STRIDE;
  v16bf a;
  ((u32x4*)&a)[0] = *(const u32x4*)(p + 16 * kh);
  ((u32x4*)&a)[1] = *(const u32x4*)(p + 16 * kh + 8);
  return a;
}
// B (32x16, 16-bit): lane n (0-15) holds column n, K=0..15; lane n+16: K=16..31
__device__ __forceinline__ v16bf load_b_frag(const __bf16* tile, int nbase, int lane) {
  const int n  = nbase + (lane & 15);
  const int kh = lane >> 4;
  const __bf16* p = tile + n * LDS_STRIDE + 16 * kh;
  v16bf b;
  ((u32x4*)&b)[0] = ((const u32x4*)p)[0];
  ((u32x4*)&b)[1] = ((const u32x4*)p)[1];
  return b;
}

// stage 16 fp32 -> 16 bf16 into LDS as two b128 stores (branchless, vectorized)
__device__ __forceinline__ void stage16_f32(__bf16* dst, const float* src) {
  const f32x4 v0 = *(const f32x4*)(src + 0);
  const f32x4 v1 = *(const f32x4*)(src + 4);
  const f32x4 v2 = *(const f32x4*)(src + 8);
  const f32x4 v3 = *(const f32x4*)(src + 12);
  u32x4 p0, p1;
  p0[0] = pkbf(v0[0], v0[1]); p0[1] = pkbf(v0[2], v0[3]);
  p0[2] = pkbf(v1[0], v1[1]); p0[3] = pkbf(v1[2], v1[3]);
  p1[0] = pkbf(v2[0], v2[1]); p1[1] = pkbf(v2[2], v2[3]);
  p1[2] = pkbf(v3[0], v3[1]); p1[3] = pkbf(v3[2], v3[3]);
  *(u32x4*)(dst + 0) = p0;
  *(u32x4*)(dst + 8) = p1;
}

// copy 16 bf16 (32B) LDS<-global
__device__ __forceinline__ void copy16_bf(__bf16* dst, const __bf16* src) {
  *(u32x4*)(dst + 0) = *(const u32x4*)(src + 0);
  *(u32x4*)(dst + 8) = *(const u32x4*)(src + 8);
}

// one K-step: 2 A-frags x 4 B-frags = 8 WMMA
__device__ __forceinline__ void mma_step(const __bf16* Ab, const __bf16* Bb,
                                         v8f acc[2][4], int wm, int wn, int lane) {
  const v16bf a0 = load_a_frag(Ab, 32 * wm, lane);
  const v16bf a1 = load_a_frag(Ab, 32 * wm + 16, lane);
#pragma unroll
  for (int ns = 0; ns < 4; ++ns) {
    const v16bf b = load_b_frag(Bb, 64 * wn + 16 * ns, lane);
    acc[0][ns] = __builtin_amdgcn_wmma_f32_16x16x32_bf16(
        false, a0, false, b, (short)0, acc[0][ns], false, false);
    acc[1][ns] = __builtin_amdgcn_wmma_f32_16x16x32_bf16(
        false, a1, false, b, (short)0, acc[1][ns], false, false);
  }
}

// ---------------- init: zero out[] and ws accumulators ----------------
__global__ void moe_init_kernel(float* __restrict__ out, int total,
                                int* __restrict__ cnt, float* __restrict__ top1,
                                float* __restrict__ colsum, float* __restrict__ ntreal) {
  const int i = blockIdx.x * blockDim.x + threadIdx.x;
  if (i < total) out[i] = 0.0f;
  if (blockIdx.x == 0 && threadIdx.x < NE) {
    cnt[threadIdx.x]    = 0;
    top1[threadIdx.x]   = 0.0f;
    colsum[threadIdx.x] = 0.0f;
    if (threadIdx.x == 0) *ntreal = 0.0f;
  }
}

// ---------------- gate: one wave32 per token ----------------
__global__ __launch_bounds__(256) void moe_gate_kernel(
    const float* __restrict__ x, const unsigned char* __restrict__ pm,
    const float* __restrict__ Wg, float* __restrict__ gate_out,
    int* __restrict__ cnt, int* __restrict__ list_tok, float* __restrict__ list_w,
    float* __restrict__ top1, float* __restrict__ colsum, float* __restrict__ ntreal) {
  const int wave = threadIdx.x >> 5, lane = threadIdx.x & 31;
  const int n = blockIdx.x * (blockDim.x >> 5) + wave;
  if (n >= NT) return;
  const float* xr = x + (size_t)n * DDIM;

  float acc[NE] = {};
  for (int i = lane; i < DDIM; i += 32) {
    const float xv = xr[i];
#pragma unroll
    for (int e = 0; e < NE; ++e) acc[e] = fmaf(xv, Wg[e * DDIM + i], acc[e]);
  }
#pragma unroll
  for (int e = 0; e < NE; ++e)
    for (int off = 16; off; off >>= 1) acc[e] += __shfl_down(acc[e], off, 32);

  if (lane == 0) {
    const bool pad = (pm[n] != 0);
    float mx = acc[0];
#pragma unroll
    for (int e = 1; e < NE; ++e) mx = fmaxf(mx, acc[e]);
    float ex[NE], s = 0.0f;
#pragma unroll
    for (int e = 0; e < NE; ++e) { ex[e] = __expf(acc[e] - mx); s += ex[e]; }
    float gv[NE];
#pragma unroll
    for (int e = 0; e < NE; ++e) {
      gv[e] = pad ? 0.0f : (ex[e] / s);
      gate_out[n * NE + e] = gv[e];
      atomicAdd(&colsum[e], gv[e]);
    }
    if (!pad) {
      atomicAdd(ntreal, 1.0f);
      int i1 = 0; float v1 = gv[0];
#pragma unroll
      for (int e = 1; e < NE; ++e) if (gv[e] > v1) { v1 = gv[e]; i1 = e; }
      int i2 = (i1 == 0) ? 1 : 0; float v2 = gv[i2];
#pragma unroll
      for (int e = 0; e < NE; ++e) if (e != i1 && gv[e] > v2) { v2 = gv[e]; i2 = e; }
      atomicAdd(&top1[i1], 1.0f);
      float ssum = v1 + v2; if (ssum == 0.0f) ssum = 1.0f;
      int s1 = atomicAdd(&cnt[i1], 1);
      list_tok[i1 * NT + s1] = n; list_w[i1 * NT + s1] = v1 / ssum;
      int s2 = atomicAdd(&cnt[i2], 1);
      list_tok[i2 * NT + s2] = n; list_w[i2 * NT + s2] = v2 / ssum;
    }
  }
}

// ---------------- finalize: prefix offsets + lb loss ----------------
__global__ void moe_finalize_kernel(const int* __restrict__ cnt, int* __restrict__ off,
                                    const float* __restrict__ top1,
                                    const float* __restrict__ colsum,
                                    const float* __restrict__ ntreal,
                                    float* __restrict__ lb_out) {
  if (threadIdx.x == 0 && blockIdx.x == 0) {
    int o = 0;
    for (int e = 0; e < NE; ++e) { off[e] = o; o += cnt[e]; }
    float ntv = *ntreal; if (ntv <= 0.0f) ntv = 1.0f;
    float lb = 0.0f;
    for (int e = 0; e < NE; ++e) lb += (top1[e] / ntv) * (colsum[e] / ntv);
    *lb_out = (float)NE * lb;
  }
}

// ---------------- FFN1: h = relu(x @ W1[e].T + b1[e]) -> bf16 scratch ----------------
// Block tile: 128 routed tokens x 128 H cols, K over D in steps of 32.
// 2x-unrolled double buffer: compile-time LDS pointers, one barrier per K-step.
__global__ __launch_bounds__(256) void moe_ffn1_kernel(
    const float* __restrict__ x, const float* __restrict__ W1, const float* __restrict__ b1,
    const int* __restrict__ cnt, const int* __restrict__ off,
    const int* __restrict__ list_tok, __bf16* __restrict__ hbuf) {
  const int e = blockIdx.z;
  const int count = cnt[e];              // >=1 if any block proceeds
  const int m0 = blockIdx.y * 128;
  if (m0 >= count) return;
  const int n0 = blockIdx.x * 128;
  const int base = off[e];
  const int t = threadIdx.x;
  const int wave = t >> 5, lane = t & 31;
  const int wm = wave & 3, wn = wave >> 2;

  __shared__ __align__(16) __bf16 At[2 * TILE_E];
  __shared__ __align__(16) __bf16 Bt[2 * TILE_E];
  __shared__ int tok[128];

  if (t < 128) {
    int m = m0 + t;
    if (m >= count) m = count - 1;       // clamp: garbage rows only feed masked outputs
    tok[t] = list_tok[e * NT + m];
  }
  __syncthreads();

  const int sr = t >> 1, sc = (t & 1) * 16;  // staging: 128 rows x 32 cols, 16 elems/thread
  const float* xrow = x + (size_t)tok[sr] * DDIM + sc;
  const float* wrow = W1 + ((size_t)e * HDIM + (size_t)(n0 + sr)) * DDIM + sc;
  __bf16* a0d = At + sr * LDS_STRIDE + sc;
  __bf16* b0d = Bt + sr * LDS_STRIDE + sc;
  __bf16* a1d = a0d + TILE_E;
  __bf16* b1d = b0d + TILE_E;

  stage16_f32(a0d, xrow);
  stage16_f32(b0d, wrow);
  __syncthreads();

  v8f acc[2][4] = {};
  for (int kk = 0; kk < DDIM; kk += 64) {
    // stage kk+32 into buffer 1, compute kk from buffer 0
    stage16_f32(a1d, xrow + kk + 32);
    stage16_f32(b1d, wrow + kk + 32);
    mma_step(At, Bt, acc, wm, wn, lane);
    __syncthreads();
    // stage kk+64 into buffer 0, compute kk+32 from buffer 1
    if (kk + 64 < DDIM) {
      stage16_f32(a0d, xrow + kk + 64);
      stage16_f32(b0d, wrow + kk + 64);
      __builtin_prefetch(wrow + kk + 96, 0, 1);
    }
    mma_step(At + TILE_E, Bt + TILE_E, acc, wm, wn, lane);
    __syncthreads();
  }

  const int ncol_l0 = 64 * wn + (lane & 15);
#pragma unroll
  for (int ai = 0; ai < 2; ++ai) {
    const int mrow_l = 32 * wm + 16 * ai + 8 * (lane >> 4);
#pragma unroll
    for (int ns = 0; ns < 4; ++ns) {
      const int col = n0 + ncol_l0 + 16 * ns;
      const float bias = b1[e * HDIM + col];
#pragma unroll
      for (int r = 0; r < 8; ++r) {
        const int m = m0 + mrow_l + r;
        if (m < count) {
          float v = acc[ai][ns][r] + bias;
          v = v > 0.0f ? v : 0.0f;
          hbuf[(size_t)(base + m) * HDIM + col] = (__bf16)v;
        }
      }
    }
  }
}

// ---------------- FFN2: out += gw * (h @ W2[e].T + b2[e]) ----------------
// Block tile: 128 pairs x 128 D cols, K over H=4096 in steps of 32, same pipeline.
__global__ __launch_bounds__(256) void moe_ffn2_kernel(
    const __bf16* __restrict__ hbuf, const float* __restrict__ W2,
    const float* __restrict__ b2, const int* __restrict__ cnt, const int* __restrict__ off,
    const int* __restrict__ list_tok, const float* __restrict__ list_w,
    float* __restrict__ out) {
  const int e = blockIdx.z;
  const int count = cnt[e];
  const int m0 = blockIdx.y * 128;
  if (m0 >= count) return;
  const int n0 = blockIdx.x * 128;
  const int base = off[e];
  const int t = threadIdx.x;
  const int wave = t >> 5, lane = t & 31;
  const int wm = wave & 3, wn = wave >> 2;

  __shared__ __align__(16) __bf16 At[2 * TILE_E];
  __shared__ __align__(16) __bf16 Bt[2 * TILE_E];
  __shared__ int   tok[128];
  __shared__ float twt[128];

  if (t < 128) {
    const int m = m0 + t;
    const int mc = (m < count) ? m : (count - 1);  // clamp for branchless staging
    tok[t] = list_tok[e * NT + mc];
    twt[t] = (m < count) ? list_w[e * NT + m] : 0.0f;
  }
  __syncthreads();

  const int sr = t >> 1, sc = (t & 1) * 16;
  const int mrc = (m0 + sr < count) ? (m0 + sr) : (count - 1);
  const __bf16* hrow = hbuf + (size_t)(base + mrc) * HDIM + sc;  // clamped: always valid
  const float* wrow = W2 + ((size_t)e * DDIM + (size_t)(n0 + sr)) * HDIM + sc;
  __bf16* a0d = At + sr * LDS_STRIDE + sc;
  __bf16* b0d = Bt + sr * LDS_STRIDE + sc;
  __bf16* a1d = a0d + TILE_E;
  __bf16* b1d = b0d + TILE_E;

  copy16_bf(a0d, hrow);
  stage16_f32(b0d, wrow);
  __syncthreads();

  v8f acc[2][4] = {};
  for (int kk = 0; kk < HDIM; kk += 64) {
    copy16_bf(a1d, hrow + kk + 32);
    stage16_f32(b1d, wrow + kk + 32);
    mma_step(At, Bt, acc, wm, wn, lane);
    __syncthreads();
    if (kk + 64 < HDIM) {
      copy16_bf(a0d, hrow + kk + 64);
      stage16_f32(b0d, wrow + kk + 64);
      __builtin_prefetch(wrow + kk + 96, 0, 1);
    }
    mma_step(At + TILE_E, Bt + TILE_E, acc, wm, wn, lane);
    __syncthreads();
  }

  const int ncol_l0 = 64 * wn + (lane & 15);
#pragma unroll
  for (int ai = 0; ai < 2; ++ai) {
    const int mrow_l = 32 * wm + 16 * ai + 8 * (lane >> 4);
#pragma unroll
    for (int ns = 0; ns < 4; ++ns) {
      const int col = n0 + ncol_l0 + 16 * ns;
      const float bias = b2[e * DDIM + col];
#pragma unroll
      for (int r = 0; r < 8; ++r) {
        const int m = m0 + mrow_l + r;
        if (m < count) {
          const int   tk = tok[mrow_l + r];
          const float wt = twt[mrow_l + r];
          atomicAdd(&out[(size_t)tk * DDIM + col], wt * (acc[ai][ns][r] + bias));
        }
      }
    }
  }
}

// ---------------- host-side launcher ----------------
extern "C" void kernel_launch(void* const* d_in, const int* in_sizes, int n_in,
                              void* d_out, int out_size, void* d_ws, size_t ws_size,
                              hipStream_t stream) {
  const float*         x   = (const float*)d_in[0];
  const unsigned char* pmk = (const unsigned char*)d_in[1];  // bool mask (1B/elem)
  const float*         Wg  = (const float*)d_in[2];
  const float*         W1  = (const float*)d_in[3];
  const float*         b1  = (const float*)d_in[4];
  const float*         W2  = (const float*)d_in[5];
  const float*         b2  = (const float*)d_in[6];

  const int OUT_N = NT * DDIM;  // 2*1024*1024
  float* out      = (float*)d_out;
  float* gate_out = out + OUT_N;
  float* lb_out   = gate_out + NT * NE;

  // Workspace layout (<= 34 MB; counters zeroed every call for determinism)
  char* ws = (char*)d_ws;
  int*    cnt      = (int*)(ws + 0);           // 8 ints: per-expert slot counters
  int*    off      = (int*)(ws + 64);          // 8 ints: exclusive prefix offsets
  float*  top1     = (float*)(ws + 128);       // 8 f: top-1 assignment counts
  float*  colsum   = (float*)(ws + 192);       // 8 f: gate_value column sums
  float*  ntreal   = (float*)(ws + 256);       // 1 f: #non-padded tokens
  int*    list_tok = (int*)(ws + 512);                       // NE*NT ints
  float*  list_w   = (float*)(ws + 512 + NE * NT * 4);       // NE*NT floats
  __bf16* hbuf     = (__bf16*)(ws + 512 + 2 * NE * NT * 4);  // (2*NT) x HDIM bf16 = 32 MB

  moe_init_kernel<<<(OUT_N + 255) / 256, 256, 0, stream>>>(out, OUT_N, cnt, top1, colsum, ntreal);
  moe_gate_kernel<<<NT / 8, 256, 0, stream>>>(x, pmk, Wg, gate_out, cnt, list_tok, list_w,
                                              top1, colsum, ntreal);
  moe_finalize_kernel<<<1, 32, 0, stream>>>(cnt, off, top1, colsum, ntreal, lb_out);
  moe_ffn1_kernel<<<dim3(HDIM / 128, NT / 128, NE), 256, 0, stream>>>(
      x, W1, b1, cnt, off, list_tok, hbuf);
  moe_ffn2_kernel<<<dim3(DDIM / 128, NT / 128, NE), 256, 0, stream>>>(
      hbuf, W2, b2, cnt, off, list_tok, list_w, out);
}
